// SDFM_34634616275110
// MI455X (gfx1250) — matrix operand order repaired and checked by reading, not compiled, not run on hardware
//
#include <hip/hip_runtime.h>
#include <math.h>

// ---------------------------------------------------------------------------
// Problem constants (from reference): N=8, IN_XC=512, IN_YC=256, OUTC=256,
// H=W=96, C2=128, K=3, NUM_LAT=5.
// ---------------------------------------------------------------------------
#define HWC   9216      // 96*96
#define WIMG  96
#define NB    8
#define OC    256
#define C2C   128
#define MT    4         // M-tiles (16 rows each) per wave: 64 output chans/wave

typedef __attribute__((ext_vector_type(16))) __bf16 v16bf;
typedef __attribute__((ext_vector_type(8)))  float  v8f;

static __device__ __forceinline__ __bf16 tobf(float f) { return (__bf16)f; }
static __device__ __forceinline__ float sigmoidf_(float s) { return 1.0f / (1.0f + expf(-s)); }

// ---------------------------------------------------------------------------
// WMMA GEMM for 1x1 conv:  Out[n, m, p] = bias[m] + sum_k W[m,k] * X[n,k,p]
// One wave per 64(M) x 16(pixels) tile: the gathered B fragment is built once
// per K-step and reused by MT=4 WMMAs (A fragments are contiguous b128 loads).
// ---------------------------------------------------------------------------
__global__ void sdfm_wmma_gemm1x1(const float* __restrict__ Wt,   // (M, Kt)
                                  const float* __restrict__ X,    // (N, Kt, HWC)
                                  const float* __restrict__ bias, // (M)
                                  float* __restrict__ Out,        // (N, M, HWC)
                                  int M, int Kt) {
  const int gwave = (blockIdx.x * blockDim.x + threadIdx.x) >> 5;
  const int lane  = threadIdx.x & 31;
  const int tilesP  = HWC / 16;                // 576
  const int groupsM = M / (16 * MT);
  const int per     = groupsM * tilesP;
  const int nb = gwave / per;
  const int t  = gwave % per;
  const int m0 = (t / tilesP) * (16 * MT);
  const int p0 = (t % tilesP) * 16;
  const int ln = lane & 15, lh = lane >> 4;
  const int am   = m0 + ln;
  const int akb  = lh * 16;
  const int pcol = p0 + ln;
  const int bkb  = lh * 16;
  const float* __restrict__ Xb = X + (size_t)nb * Kt * HWC;

  v8f acc[MT];
#pragma unroll
  for (int mt = 0; mt < MT; ++mt) acc[mt] = (v8f){};

  for (int k0 = 0; k0 < Kt; k0 += 32) {
    v16bf b;
#pragma unroll
    for (int i = 0; i < 16; ++i) {
      b[i] = tobf(Xb[(size_t)(k0 + bkb + i) * HWC + pcol]);
    }
#pragma unroll
    for (int mt = 0; mt < MT; ++mt) {
      const float* wr = Wt + (size_t)(am + mt * 16) * Kt + k0;
      __builtin_prefetch(wr + 64, 0, 1);       // global_prefetch_b8 on weight stream
      v16bf a;
#pragma unroll
      for (int i = 0; i < 8; ++i) {
        a[i]     = tobf(wr[akb + i]);
        a[8 + i] = tobf(wr[akb + 8 + i]);
      }
      acc[mt] = __builtin_amdgcn_wmma_f32_16x16x32_bf16(false, a, false, b,
                                                        (short)0, acc[mt], false, false);
    }
  }
#pragma unroll
  for (int mt = 0; mt < MT; ++mt) {
#pragma unroll
    for (int r = 0; r < 8; ++r) {
      const int m = m0 + mt * 16 + r + lh * 8;
      Out[((size_t)nb * M + m) * HWC + pcol] = acc[mt][r] + bias[m];
    }
  }
}

// ---------------------------------------------------------------------------
// WMMA implicit GEMM for 3x3 conv (pad=1):
//   Out[n, m, p] = epilogue( sum_{ci,t} W[m, ci*9+t] * Inpad[n, ci, p+off(t)] )
// Input channels split across two tensors (for the concat in the fuse conv):
//   ci <  C0 -> In0 (batch stride sN0);  ci >= C0 -> In1 (batch stride sN1).
// The expensive gathered B fragment is reused across MT=4 WMMAs per K-step.
// Epilogue: optional bias; optional bn(scale/shift)+PReLU.
// ---------------------------------------------------------------------------
__global__ void sdfm_wmma_conv3x3(const float* __restrict__ Wt,    // (M, Cin*9)
                                  const float* __restrict__ bias,  // (M) or null
                                  const float* __restrict__ gma,   // bn gamma or null
                                  const float* __restrict__ bta,   // bn beta
                                  const float* __restrict__ alp,   // prelu alpha
                                  const float* __restrict__ In0, long sN0, int C0,
                                  const float* __restrict__ In1, long sN1,
                                  int Cin, int M,
                                  float* __restrict__ Out) {
  const int gwave = (blockIdx.x * blockDim.x + threadIdx.x) >> 5;
  const int lane  = threadIdx.x & 31;
  const int tilesP  = HWC / 16;
  const int groupsM = M / (16 * MT);
  const int per     = groupsM * tilesP;
  const int nb = gwave / per;
  const int t  = gwave % per;
  const int m0 = (t / tilesP) * (16 * MT);
  const int p0 = (t % tilesP) * 16;
  const int ln = lane & 15, lh = lane >> 4;
  const int am   = m0 + ln;
  const int akb  = lh * 16;
  const int pcol = p0 + ln;
  const int py = pcol / WIMG, px = pcol % WIMG;
  const int bkb = lh * 16;
  const int Ktot = Cin * 9;
  const float* __restrict__ b0 = In0 + (size_t)nb * sN0;
  const float* __restrict__ b1 = In1 + (size_t)nb * sN1;

  v8f acc[MT];
#pragma unroll
  for (int mt = 0; mt < MT; ++mt) acc[mt] = (v8f){};

  for (int k0 = 0; k0 < Ktot; k0 += 32) {
    v16bf b;
#pragma unroll
    for (int i = 0; i < 16; ++i) {
      const int k  = k0 + bkb + i;
      const int ci = k / 9;
      const int tt = k - ci * 9;
      const int ty = tt / 3;
      const int dy = ty - 1;
      const int dx = (tt - ty * 3) - 1;
      const int yy = py + dy, xx = px + dx;
      const float* src = (ci < C0) ? (b0 + (size_t)ci * HWC)
                                   : (b1 + (size_t)(ci - C0) * HWC);
      const float v = ((unsigned)yy < (unsigned)WIMG && (unsigned)xx < (unsigned)WIMG)
                          ? src[yy * WIMG + xx] : 0.0f;
      b[i] = tobf(v);
    }
#pragma unroll
    for (int mt = 0; mt < MT; ++mt) {
      const float* wr = Wt + (size_t)(am + mt * 16) * Ktot + k0;
      __builtin_prefetch(wr + 64, 0, 1);
      v16bf a;
#pragma unroll
      for (int i = 0; i < 8; ++i) {
        a[i]     = tobf(wr[akb + i]);
        a[8 + i] = tobf(wr[akb + 8 + i]);
      }
      acc[mt] = __builtin_amdgcn_wmma_f32_16x16x32_bf16(false, a, false, b,
                                                        (short)0, acc[mt], false, false);
    }
  }
#pragma unroll
  for (int mt = 0; mt < MT; ++mt) {
#pragma unroll
    for (int r = 0; r < 8; ++r) {
      const int m = m0 + mt * 16 + r + lh * 8;
      float v = acc[mt][r];
      if (bias) v += bias[m];
      if (gma) {                       // bn2 scale/shift + PReLU (fuse epilogue)
        v = v * gma[m] + bta[m];
        v = (v >= 0.0f) ? v : alp[m] * v;
      }
      Out[((size_t)nb * M + m) * HWC + pcol] = v;
    }
  }
}

// ---------------------------------------------------------------------------
// pool_to_3x3: one wave per (n, c, region); region = 32x32 mean.
// ---------------------------------------------------------------------------
__global__ void sdfm_pool3x3(const float* __restrict__ src, long strideN, int C,
                             float* __restrict__ dst /* (N, C, 9) */) {
  const int wid  = (blockIdx.x * blockDim.x + threadIdx.x) >> 5;
  const int lane = threadIdx.x & 31;
  const int r = wid % 9;
  const int c = (wid / 9) % C;
  const int n = wid / (9 * C);
  const int ry = r / 3, rx = r % 3;
  const float* p = src + (size_t)n * strideN + (size_t)c * HWC + ry * 32 * WIMG + rx * 32;
  float s = 0.0f;
#pragma unroll 4
  for (int i = 0; i < 32; ++i) s += p[i * WIMG + lane];
  for (int o = 16; o > 0; o >>= 1) s += __shfl_down(s, o, 32);
  if (lane == 0) dst[wid] = s * (1.0f / 1024.0f);
}

// ---------------------------------------------------------------------------
// Attention chain + kc: one thread per (n, c).
//   lat = relu(bn(gl @ w_ce^T)); att = sigmoid(lat @ w_gd^T)
//   kc  = (b_kc + w_kc @ ply) * att
// ---------------------------------------------------------------------------
__global__ void sdfm_att_kc(const float* __restrict__ gl, const float* __restrict__ ply,
                            const float* __restrict__ w_ce, const float* __restrict__ w_gd,
                            const float* __restrict__ bn_g, const float* __restrict__ bn_b,
                            const float* __restrict__ w_kc, const float* __restrict__ b_kc,
                            float* __restrict__ kc) {
  const int idx = blockIdx.x * blockDim.x + threadIdx.x;   // N*C2 threads
  const int n = idx / C2C, c = idx % C2C;
  float glv[9];
#pragma unroll
  for (int j = 0; j < 9; ++j) glv[j] = gl[((size_t)n * C2C + c) * 9 + j];
  float lat[5];
#pragma unroll
  for (int l = 0; l < 5; ++l) {
    float s = 0.0f;
#pragma unroll
    for (int j = 0; j < 9; ++j) s += w_ce[l * 9 + j] * glv[j];
    s = s * bn_g[c] + bn_b[c];
    lat[l] = fmaxf(s, 0.0f);
  }
  float att[9];
#pragma unroll
  for (int j = 0; j < 9; ++j) {
    float s = 0.0f;
#pragma unroll
    for (int l = 0; l < 5; ++l) s += w_gd[j * 5 + l] * lat[l];
    att[j] = sigmoidf_(s);
  }
  float accv[9];
#pragma unroll
  for (int j = 0; j < 9; ++j) accv[j] = b_kc[c];
  for (int cp = 0; cp < C2C; ++cp) {
    const float wv = w_kc[c * C2C + cp];
    const float* pr = ply + ((size_t)n * C2C + cp) * 9;
#pragma unroll
    for (int j = 0; j < 9; ++j) accv[j] += wv * pr[j];
  }
#pragma unroll
  for (int j = 0; j < 9; ++j) kc[((size_t)n * C2C + c) * 9 + j] = accv[j] * att[j];
}

// ---------------------------------------------------------------------------
// channel_after: res[:, c<128] = sum_j patches(x1)[c,j] * kc[c,j]
// One block per (n, c); threads stride pixels.
// ---------------------------------------------------------------------------
__global__ void sdfm_channel_after(const float* __restrict__ xd,
                                   const float* __restrict__ kc,
                                   float* __restrict__ res) {
  const int n = blockIdx.x / C2C, c = blockIdx.x % C2C;
  const float* x1 = xd + ((size_t)n * OC + c) * HWC;
  float* o = res + ((size_t)n * OC + c) * HWC;
  float k9[9];
#pragma unroll
  for (int j = 0; j < 9; ++j) k9[j] = kc[((size_t)n * C2C + c) * 9 + j];
  for (int p = threadIdx.x; p < HWC; p += blockDim.x) {
    const int y = p / WIMG, x = p % WIMG;
    float s = 0.0f;
#pragma unroll
    for (int j = 0; j < 9; ++j) {
      const int yy = y + j / 3 - 1, xx = x + j % 3 - 1;
      if ((unsigned)yy < (unsigned)WIMG && (unsigned)xx < (unsigned)WIMG)
        s += x1[yy * WIMG + xx] * k9[j];
    }
    o[p] = s;
  }
}

// ---------------------------------------------------------------------------
// ksp = conv(x2, w_sp1 (1,128,3,3), pad=1). One thread per (n, pixel).
// ---------------------------------------------------------------------------
__global__ void sdfm_ksp(const float* __restrict__ xd,
                         const float* __restrict__ w_sp1,
                         float* __restrict__ ksp) {
  __shared__ float wsm[C2C * 9];
  for (int i = threadIdx.x; i < C2C * 9; i += blockDim.x) wsm[i] = w_sp1[i];
  __syncthreads();
  const int n = blockIdx.x / 36;
  const int p = (blockIdx.x % 36) * 256 + threadIdx.x;
  const int y = p / WIMG, x = p % WIMG;
  const float* x2 = xd + ((size_t)n * OC + C2C) * HWC;
  float s = 0.0f;
  for (int c = 0; c < C2C; ++c) {
    const float* pl = x2 + (size_t)c * HWC;
    const float* wc = wsm + c * 9;
#pragma unroll
    for (int j = 0; j < 9; ++j) {
      const int yy = y + j / 3 - 1, xx = x + j % 3 - 1;
      if ((unsigned)yy < (unsigned)WIMG && (unsigned)xx < (unsigned)WIMG)
        s += pl[yy * WIMG + xx] * wc[j];
    }
  }
  ksp[(size_t)n * HWC + p] = s;
}

// ---------------------------------------------------------------------------
// avg/max of 9 zero-padded neighbors of ksp -> am (N, 2, HWC)
// ---------------------------------------------------------------------------
__global__ void sdfm_avgmx(const float* __restrict__ ksp, float* __restrict__ am) {
  const int idx = blockIdx.x * blockDim.x + threadIdx.x;
  const int n = idx / HWC, p = idx % HWC;
  const int y = p / WIMG, x = p % WIMG;
  const float* kp = ksp + (size_t)n * HWC;
  float s = 0.0f, mx = -1e30f;
#pragma unroll
  for (int j = 0; j < 9; ++j) {
    const int yy = y + j / 3 - 1, xx = x + j % 3 - 1;
    const float v = ((unsigned)yy < (unsigned)WIMG && (unsigned)xx < (unsigned)WIMG)
                        ? kp[yy * WIMG + xx] : 0.0f;   // padded zeros participate
    s += v;
    mx = fmaxf(mx, v);
  }
  am[((size_t)n * 2 + 0) * HWC + p] = s * (1.0f / 9.0f);
  am[((size_t)n * 2 + 1) * HWC + p] = mx;
}

// ---------------------------------------------------------------------------
// sp = sigmoid(conv([avg, mx], w_sp (1,2,3,3), pad=1))
// ---------------------------------------------------------------------------
__global__ void sdfm_sp(const float* __restrict__ am, const float* __restrict__ w_sp,
                        float* __restrict__ sp) {
  const int idx = blockIdx.x * blockDim.x + threadIdx.x;
  const int n = idx / HWC, p = idx % HWC;
  const int y = p / WIMG, x = p % WIMG;
  float s = 0.0f;
#pragma unroll
  for (int c = 0; c < 2; ++c) {
    const float* pl = am + ((size_t)n * 2 + c) * HWC;
#pragma unroll
    for (int j = 0; j < 9; ++j) {
      const int yy = y + j / 3 - 1, xx = x + j % 3 - 1;
      if ((unsigned)yy < (unsigned)WIMG && (unsigned)xx < (unsigned)WIMG)
        s += pl[yy * WIMG + xx] * w_sp[c * 9 + j];
    }
  }
  sp[(size_t)n * HWC + p] = sigmoidf_(s);
}

// ---------------------------------------------------------------------------
// ks2: ks[n, j, p] = b_ks2[j] + sum_c w_ks2[j, c] * yk1[n, c, p]
// ---------------------------------------------------------------------------
__global__ void sdfm_ks2(const float* __restrict__ yk1, const float* __restrict__ w_ks2,
                         const float* __restrict__ b_ks2, float* __restrict__ ks) {
  const int idx = blockIdx.x * blockDim.x + threadIdx.x;
  const int n = idx / HWC, p = idx % HWC;
  float kv[9];
#pragma unroll
  for (int j = 0; j < 9; ++j) kv[j] = b_ks2[j];
  for (int c = 0; c < C2C; ++c) {
    const float v = yk1[((size_t)n * C2C + c) * HWC + p];
#pragma unroll
    for (int j = 0; j < 9; ++j) kv[j] += w_ks2[j * C2C + c] * v;
  }
#pragma unroll
  for (int j = 0; j < 9; ++j) ks[((size_t)n * 9 + j) * HWC + p] = kv[j];
}

// ---------------------------------------------------------------------------
// spatial_after with the reference's *reinterpreting* reshape:
//   flat(c2*9, hw) order reinterpreted as (h, w, c2, 9). For output pixel p,
//   element (c', j) reads flat index f = p*1152 + c'*9 + j decoded back as
//   (c = f/82944, jj = (f%82944)/9216, pp = f%9216) into patches(x2).
// Incremental walk -> no divides in the inner loop. Writes res channels 128..255.
// ---------------------------------------------------------------------------
__global__ void sdfm_spatial(const float* __restrict__ xd, const float* __restrict__ ks,
                             const float* __restrict__ sp, float* __restrict__ res) {
  const int idx = blockIdx.x * blockDim.x + threadIdx.x;
  const int n = idx / HWC, p = idx % HWC;
  const float spv = sp[(size_t)n * HWC + p];
  float w9[9];
#pragma unroll
  for (int j = 0; j < 9; ++j) w9[j] = ks[((size_t)n * 9 + j) * HWC + p] * spv;

  const float* x2 = xd + ((size_t)n * OC + C2C) * HWC;
  float* resb = res + ((size_t)n * OC + C2C) * HWC;

  // decode starting flat index f = p * 1152  (1152 = C2*9, 82944 = 9*HWC)
  int f   = p * 1152;
  int c   = f / 82944;
  int rem = f - c * 82944;
  int jj  = rem / HWC;
  int pp  = rem - jj * HWC;
  int ppy = pp / WIMG;
  int ppx = pp - ppy * WIMG;

  float accv = 0.0f;
  int j = 0, cp = 0;
  for (int it = 0; it < 1152; ++it) {
    const int dy = jj / 3 - 1, dx = jj % 3 - 1;
    const int yy = ppy + dy, xx = ppx + dx;
    const float v = ((unsigned)yy < (unsigned)WIMG && (unsigned)xx < (unsigned)WIMG)
                        ? x2[(size_t)c * HWC + yy * WIMG + xx] : 0.0f;
    accv += v * w9[j];
    if (++j == 9) {
      resb[(size_t)cp * HWC + p] = accv;
      accv = 0.0f; j = 0; ++cp;
    }
    if (++ppx == WIMG) { ppx = 0; ++ppy; }
    if (ppy == WIMG)   { ppy = 0; ++jj; if (jj == 9) { jj = 0; ++c; } }
  }
}

// ---------------------------------------------------------------------------
// Orchestration
// ---------------------------------------------------------------------------
extern "C" void kernel_launch(void* const* d_in, const int* in_sizes, int n_in,
                              void* d_out, int out_size, void* d_ws, size_t ws_size,
                              hipStream_t stream) {
  const float* x       = (const float*)d_in[0];
  const float* y       = (const float*)d_in[1];
  const float* w_down  = (const float*)d_in[2];
  const float* b_down  = (const float*)d_in[3];
  const float* w_ce    = (const float*)d_in[4];
  const float* w_gd    = (const float*)d_in[5];
  const float* bn_g    = (const float*)d_in[6];
  const float* bn_b    = (const float*)d_in[7];
  const float* w_kc    = (const float*)d_in[8];
  const float* b_kc    = (const float*)d_in[9];
  const float* w_ks1   = (const float*)d_in[10];
  const float* b_ks1   = (const float*)d_in[11];
  const float* w_ks2   = (const float*)d_in[12];
  const float* b_ks2   = (const float*)d_in[13];
  const float* w_sp1   = (const float*)d_in[14];
  const float* w_sp    = (const float*)d_in[15];
  const float* w_fb    = (const float*)d_in[16];
  const float* b_fb    = (const float*)d_in[17];
  const float* w_fuse  = (const float*)d_in[18];
  const float* bn2_g   = (const float*)d_in[19];
  const float* bn2_b   = (const float*)d_in[20];
  const float* prelu_a = (const float*)d_in[21];
  float* out = (float*)d_out;
  float* ws  = (float*)d_ws;

  const size_t NCHW = (size_t)NB * OC * HWC;    // 18,874,368 floats
  float* xd  = ws;                              // down-conv output (N,256,HW)
  float* res = ws + NCHW;                       // [channel_after | spatial_after]
  float* r1  = ws + 2 * NCHW;                   // feedback conv output
  float* yk1 = ws + 3 * NCHW;                   // ks1 output (N,128,HW)
  float* ks  = yk1 + (size_t)NB * C2C * HWC;    // (N,9,HW)
  float* ksp = ks  + (size_t)NB * 9 * HWC;      // (N,HW)
  float* am  = ksp + (size_t)NB * HWC;          // (N,2,HW)
  float* sp  = am  + (size_t)NB * 2 * HWC;      // (N,HW)
  float* gl  = sp  + (size_t)NB * HWC;          // (N,128,9)
  float* ply = gl  + (size_t)NB * C2C * 9;
  float* kc  = ply + (size_t)NB * C2C * 9;

  const dim3 blk(256);
  const long sN256 = (long)OC * HWC;            // batch stride of a 256-ch tensor

  // waves = NB * (M/(16*MT)) * 576 ; blocks = waves/8 (256 thr = 8 waves)
  const int blk_m256 = NB * (OC  / (16 * MT)) * (HWC / 16) / 8;  // 2304
  const int blk_m128 = NB * (C2C / (16 * MT)) * (HWC / 16) / 8;  // 1152

  // 1) x = conv1x1(x, w_down) + b_down           [WMMA, M=256 K=512]
  sdfm_wmma_gemm1x1<<<blk_m256, blk, 0, stream>>>(w_down, x, b_down, xd, OC, 512);

  // 2) yk1 = conv3x3(y2, w_ks1) + b_ks1          [WMMA, M=128 K=1152]
  sdfm_wmma_conv3x3<<<blk_m128, blk, 0, stream>>>(
      w_ks1, b_ks1, nullptr, nullptr, nullptr,
      y + (size_t)C2C * HWC, sN256, C2C,
      y + (size_t)C2C * HWC, sN256, C2C, C2C, yk1);

  // 3) pooled x1 and y1
  sdfm_pool3x3<<<1152, blk, 0, stream>>>(xd, sN256, C2C, gl);
  sdfm_pool3x3<<<1152, blk, 0, stream>>>(y,  sN256, C2C, ply);

  // 4) attention + kc
  sdfm_att_kc<<<4, blk, 0, stream>>>(gl, ply, w_ce, w_gd, bn_g, bn_b, w_kc, b_kc, kc);

  // 5) channel branch -> res[:, 0:128]
  sdfm_channel_after<<<NB * C2C, blk, 0, stream>>>(xd, kc, res);

  // 6-8) spatial gate: ksp -> avg/max -> sp
  sdfm_ksp<<<NB * 36, blk, 0, stream>>>(xd, w_sp1, ksp);
  sdfm_avgmx<<<NB * 36, blk, 0, stream>>>(ksp, am);
  sdfm_sp<<<NB * 36, blk, 0, stream>>>(am, w_sp, sp);

  // 9) ks = conv1x1(yk1, w_ks2) + b_ks2 (9 outputs)
  sdfm_ks2<<<NB * 36, blk, 0, stream>>>(yk1, w_ks2, b_ks2, ks);

  // 10) spatial branch -> res[:, 128:256]
  sdfm_spatial<<<NB * 36, blk, 0, stream>>>(xd, ks, sp, res);

  // 11) r1 = conv3x3(res, w_fb) + b_fb           [WMMA, M=256 K=2304]
  sdfm_wmma_conv3x3<<<blk_m256, blk, 0, stream>>>(
      w_fb, b_fb, nullptr, nullptr, nullptr,
      res, sN256, OC, res, sN256, OC, OC, r1);

  // 12) o = prelu(bn2(conv3x3(concat(xd, r1), w_fuse)))  [WMMA, M=256 K=4608]
  sdfm_wmma_conv3x3<<<blk_m256, blk, 0, stream>>>(
      w_fuse, nullptr, bn2_g, bn2_b, prelu_a,
      xd, sN256, OC, r1, sN256, OC + OC, OC, out);

  (void)in_sizes; (void)n_in; (void)out_size; (void)ws_size;
}
